// MultiheadAttention_59957743452610
// MI455X (gfx1250) — compile-verified, matrix operand order checked by reading, not run on hardware
//
#include <hip/hip_runtime.h>
#include <hip/hip_bf16.h>

// ---------------------------------------------------------------------------
// MultiheadAttention (T=1024, B=8, E=1024, H=16, DH=64) for MI455X / gfx1250.
//   * all matmuls on v_wmma_f32_16x16x32_bf16 (bf16 in, f32 accumulate)
//   * GEMM k-loops: explicit 2-stage register double buffer (no rotation movs)
//   * flash attention: K/V tiles ping-pong double-buffered in LDS via the
//     Tensor Data Mover (tensor_load_to_lds overlapped with compute,
//     s_wait_tensorcnt), ds_swizzle row reductions, LDS P relayout,
//     global_prefetch_b8 for upcoming tiles.
// ---------------------------------------------------------------------------

typedef __bf16 bf16_t;
typedef __attribute__((ext_vector_type(16))) __bf16 v16bf;
typedef __attribute__((ext_vector_type(8)))  float  v8f;
typedef __attribute__((ext_vector_type(4)))  unsigned int u32x4;
typedef __attribute__((ext_vector_type(4)))  int  i32x4;
typedef __attribute__((ext_vector_type(8)))  int  i32x8;

#define WMMA_BF16(a, b, c) \
    __builtin_amdgcn_wmma_f32_16x16x32_bf16(false, (a), false, (b), 0, (c), false, false)

constexpr int T_  = 1024;
constexpr int B_  = 8;
constexpr int E_  = 1024;
constexpr int H_  = 16;
constexpr int DH_ = 64;
constexpr int M_  = T_ * B_;          // 8192 tokens
constexpr float SCALING = 0.125f;     // DH^-0.5
constexpr float MASKED_BIAS = -10000.0f;

#if __has_builtin(__builtin_amdgcn_tensor_load_to_lds) && \
    __has_builtin(__builtin_amdgcn_s_wait_tensorcnt)
#define USE_TDM 1
#else
#define USE_TDM 0
#endif

// ---- ds_swizzle-based 16-lane XOR max reduction (wave32 lane permute) ------
template <int PAT>
__device__ __forceinline__ float swzf(float v) {
    int i = __builtin_amdgcn_ds_swizzle(__builtin_bit_cast(int, v), PAT);
    return __builtin_bit_cast(float, i);
}
__device__ __forceinline__ float rowmax16(float v) {
    // group-of-32 swizzle: pattern = (xor_mask<<10) | and_mask(0x1f)
    v = fmaxf(v, swzf<0x041f>(v));   // xor 1
    v = fmaxf(v, swzf<0x081f>(v));   // xor 2
    v = fmaxf(v, swzf<0x101f>(v));   // xor 4
    v = fmaxf(v, swzf<0x201f>(v));   // xor 8
    return v;
}

#if USE_TDM
// ---- Tensor Data Mover: stage a 2D bf16 tile (tile_h rows x tile_w elems)
// from global memory (row stride = stride_elems) into LDS at lds_off.
__device__ __forceinline__ void tdm_load_2d(unsigned lds_off, const void* gptr,
                                            unsigned tile_w, unsigned tile_h,
                                            unsigned stride_elems)
{
    const unsigned long long ga = (unsigned long long)(uintptr_t)gptr;
    u32x4 g0;
    g0[0] = 1u;                                        // count=1 user D#
    g0[1] = lds_off;                                   // lds_addr (bytes)
    g0[2] = (unsigned)(ga & 0xffffffffull);            // global_addr lo
    g0[3] = (unsigned)((ga >> 32) & 0x01ffffffull)     // global_addr hi
          | (2u << 30);                                // type = 2 ("image")
    i32x8 g1;
    g1[0] = (1 << 16);                                 // data_size=1 (2 bytes)
    g1[1] = (int)((tile_w & 0xffffu) << 16);           // tensor_dim0 lo16
    g1[2] = (int)((tile_w >> 16) & 0xffffu)            // tensor_dim0 hi16
          | (int)((tile_h & 0xffffu) << 16);           // tensor_dim1 lo16
    g1[3] = (int)((tile_h >> 16) & 0xffffu)            // tensor_dim1 hi16
          | (int)((tile_w & 0xffffu) << 16);           // tile_dim0
    g1[4] = (int)(tile_h & 0xffffu);                   // tile_dim1 (tile_dim2=0)
    g1[5] = (int)stride_elems;                         // tensor_dim0_stride lo32
    g1[6] = 0;                                         // stride0 hi / stride1 lo
    g1[7] = 0;                                         // stride1 hi
    i32x4 gz = {0, 0, 0, 0};
#if __clang_major__ >= 23
    i32x8 gz8 = {0, 0, 0, 0, 0, 0, 0, 0};
    __builtin_amdgcn_tensor_load_to_lds(g0, g1, gz, gz, gz8, 0);
#else
    __builtin_amdgcn_tensor_load_to_lds(g0, g1, gz, gz, 0);
#endif
}
#endif

// ---------------------------------------------------------------------------
// f32 -> bf16 conversion (bandwidth trivial; working set then fits in L2)
// ---------------------------------------------------------------------------
__global__ void cvt_f32_bf16(const float* __restrict__ in,
                             bf16_t* __restrict__ out, int n) {
    int i = blockIdx.x * blockDim.x + threadIdx.x;
    int stride = gridDim.x * blockDim.x;
    for (; i < n; i += stride) out[i] = (bf16_t)in[i];
}

// ---------------------------------------------------------------------------
// Shared GEMM core: one wave computes a 16x64 tile of C = A * W^T with an
// explicit 2-stage register pipeline (A/B fragment sets alternate; loads for
// stage B issue while stage A's WMMAs execute -> partial loadcnt waits, and
// no register-rotation copies).
// ---------------------------------------------------------------------------
__device__ __forceinline__ void gemm_16x64_core(
    const bf16_t* __restrict__ arow, const bf16_t* const wp[4], v8f acc[4])
{
    v16bf afA, bfA[4], afB, bfB[4];
    afA = *(const v16bf*)(arow);
#pragma unroll
    for (int ns = 0; ns < 4; ++ns) bfA[ns] = *(const v16bf*)(wp[ns]);

    int k = 0;
    for (; k + 64 < E_; k += 64) {
        afB = *(const v16bf*)(arow + k + 32);
#pragma unroll
        for (int ns = 0; ns < 4; ++ns) bfB[ns] = *(const v16bf*)(wp[ns] + k + 32);
#pragma unroll
        for (int ns = 0; ns < 4; ++ns) acc[ns] = WMMA_BF16(afA, bfA[ns], acc[ns]);
        afA = *(const v16bf*)(arow + k + 64);
#pragma unroll
        for (int ns = 0; ns < 4; ++ns) bfA[ns] = *(const v16bf*)(wp[ns] + k + 64);
#pragma unroll
        for (int ns = 0; ns < 4; ++ns) acc[ns] = WMMA_BF16(afB, bfB[ns], acc[ns]);
    }
    // k == E_-64: chunks k (in A set) and k+32 remain
    afB = *(const v16bf*)(arow + k + 32);
#pragma unroll
    for (int ns = 0; ns < 4; ++ns) bfB[ns] = *(const v16bf*)(wp[ns] + k + 32);
#pragma unroll
    for (int ns = 0; ns < 4; ++ns) acc[ns] = WMMA_BF16(afA, bfA[ns], acc[ns]);
#pragma unroll
    for (int ns = 0; ns < 4; ++ns) acc[ns] = WMMA_BF16(afB, bfB[ns], acc[ns]);
}

// ---------------------------------------------------------------------------
// QKV GEMM: C[M,3E] = A[M,E] * W[3E,E]^T + bias -> head-major Q/K, transposed V.
// ---------------------------------------------------------------------------
__global__ void __launch_bounds__(256) gemm_qkv(
    const bf16_t* __restrict__ A,     // [M, E]
    const bf16_t* __restrict__ W,     // [3E, E]
    const float*  __restrict__ bias,  // [3E]
    bf16_t* __restrict__ Qh, bf16_t* __restrict__ Kh, bf16_t* __restrict__ Vt)
{
    const int lane  = threadIdx.x & 31;
    const int wave  = threadIdx.x >> 5;
    const int lmod  = lane & 15;
    const int lhalf = lane >> 4;

    const int mtile = blockIdx.x * 8 + wave;   // 0..511
    const int mbase = mtile * 16;
    const int nbase = blockIdx.y * 64;         // 0..3008

    v8f acc[4];
    v8f z = {};
    acc[0] = z; acc[1] = z; acc[2] = z; acc[3] = z;

    const bf16_t* arow = A + (size_t)(mbase + lmod) * E_ + lhalf * 16;
    const bf16_t* wp[4];
#pragma unroll
    for (int ns = 0; ns < 4; ++ns)
        wp[ns] = W + (size_t)(nbase + ns * 16 + lmod) * E_ + lhalf * 16;

    gemm_16x64_core(arow, wp, acc);

#pragma unroll
    for (int ns = 0; ns < 4; ++ns) {
        const int f0   = nbase + ns * 16;
        const int part = f0 >> 10;            // uniform: 0=Q 1=K 2=V
        const int e0   = f0 & 1023;
        const int h    = e0 >> 6;             // uniform
        const int dlo  = e0 & 63;             // uniform (0/16/32/48)
        const int d    = dlo + lmod;
        const float bv = bias[f0 + lmod];
#pragma unroll
        for (int r = 0; r < 8; ++r) {
            const int m = mbase + r + lhalf * 8;
            const int t = m >> 3;
            const int b = m & 7;
            const int nh = b * H_ + h;
            float v = acc[ns][r] + bv;
            if (part == 0) {
                Qh[((size_t)nh * T_ + t) * DH_ + d] = (bf16_t)(v * SCALING);
            } else if (part == 1) {
                Kh[((size_t)nh * T_ + t) * DH_ + d] = (bf16_t)v;
            } else {
                Vt[((size_t)nh * DH_ + d) * T_ + t] = (bf16_t)v;
            }
        }
    }
}

// ---------------------------------------------------------------------------
// Fused causal flash attention. Block = 4 waves = 4 Q tiles of ONE head.
// K/V 32-key tiles ping-pong in LDS: TDM DMA for tile i+1 overlaps the
// 9 WMMAs of tile i; one workgroup barrier per step publishes the tile.
// ---------------------------------------------------------------------------
__global__ void __launch_bounds__(128) attn_fused(
    const bf16_t* __restrict__ Qh, const bf16_t* __restrict__ Kh,
    const bf16_t* __restrict__ Vt, bf16_t* __restrict__ AO)
{
    __shared__ __align__(128) bf16_t kbuf[2][32][64];  // K rows  [key][d]
    __shared__ __align__(128) bf16_t vbuf[2][64][32];  // V^T     [d][key]
    __shared__ __align__(128) bf16_t plds[4][16][32];  // per-wave P tile

    const int lane  = threadIdx.x & 31;
    const int wave  = threadIdx.x >> 5;
    const int lmod  = lane & 15;
    const int lhalf = lane >> 4;

    const int nh = blockIdx.x;            // 0..127  (= b*H + h)
    const int b  = nh >> 4;
    const int h  = nh & 15;
    const int iq = blockIdx.y * 4 + wave; // 0..63
    const int qbase = iq * 16;

    const bf16_t* Qn = Qh + (size_t)nh * T_ * DH_;
    const bf16_t* Kn = Kh + (size_t)nh * T_ * DH_;
    const bf16_t* Vn = Vt + (size_t)nh * DH_ * T_;

    // Q fragments: frag f covers d in [32f, 32f+32)
    const v16bf qf0 = *(const v16bf*)(Qn + (size_t)(qbase + lmod) * DH_ + 0  + lhalf * 16);
    const v16bf qf1 = *(const v16bf*)(Qn + (size_t)(qbase + lmod) * DH_ + 32 + lhalf * 16);

    v16bf ones;
#pragma unroll
    for (int i = 0; i < 16; ++i) ones[i] = (bf16_t)1.0f;

    v8f o[4];
    v8f z = {};
    o[0] = z; o[1] = z; o[2] = z; o[3] = z;

    float mrow[8], lrow[8], scl[8];
#pragma unroll
    for (int r = 0; r < 8; ++r) { mrow[r] = -1e30f; lrow[r] = 0.0f; }

    const int jend = qbase + 16;              // this wave's causal key bound
    const int jmax = blockIdx.y * 64 + 64;    // block-wide key bound (wave 3)

#if USE_TDM
    // prologue: launch DMA for tile 0 into buffer 0
    if (wave == 0) {
        tdm_load_2d((unsigned)(uintptr_t)&kbuf[0][0][0], Kn, DH_, 32, DH_);
        tdm_load_2d((unsigned)(uintptr_t)&vbuf[0][0][0], Vn, 32, DH_, T_);
    }
#endif

    for (int jb = 0; jb < jmax; jb += 32) {
        const int cur = (jb >> 5) & 1;
#if USE_TDM
        if (wave == 0) __builtin_amdgcn_s_wait_tensorcnt(0);
        __syncthreads();   // tile `cur` ready; spare buffer fully consumed
        if (wave == 0 && jb + 32 < jmax) {   // overlap next DMA with compute
            tdm_load_2d((unsigned)(uintptr_t)&kbuf[cur ^ 1][0][0],
                        Kn + (size_t)(jb + 32) * DH_, DH_, 32, DH_);
            tdm_load_2d((unsigned)(uintptr_t)&vbuf[cur ^ 1][0][0],
                        Vn + jb + 32, 32, DH_, T_);
        }
#else
        __syncthreads();   // spare buffer fully consumed
        {   // fallback: 128 threads copy 4KB + 4KB (32B vector chunks)
            const int c = threadIdx.x * 16;            // K: flat elem index
            const int kr = c >> 6, kc = c & 63;
            *(v16bf*)&kbuf[cur][kr][kc] =
                *(const v16bf*)(Kn + (size_t)(jb + kr) * DH_ + kc);
            const int vr = c >> 5, vc = c & 31;
            *(v16bf*)&vbuf[cur][vr][vc] =
                *(const v16bf*)(Vn + (size_t)vr * T_ + jb + vc);
        }
        __syncthreads();   // tile visible to all waves
#endif

        // prefetch the tile after next into L2 (global_prefetch_b8)
        __builtin_prefetch(Kn + (size_t)(jb + 64 + lane) * DH_, 0, 0);
        __builtin_prefetch(Vn + (size_t)(lane * 2) * T_ + jb + 64, 0, 0);

        if (jb < jend) {   // uniform per wave: causal skip
            // ---- S = Q * K^T for 32 keys (two 16-key subtiles) ----
            v8f s0 = z, s1 = z;
            {
                v16bf k00 = *(const v16bf*)&kbuf[cur][ 0 + lmod][ 0 + lhalf * 16];
                v16bf k01 = *(const v16bf*)&kbuf[cur][ 0 + lmod][32 + lhalf * 16];
                v16bf k10 = *(const v16bf*)&kbuf[cur][16 + lmod][ 0 + lhalf * 16];
                v16bf k11 = *(const v16bf*)&kbuf[cur][16 + lmod][32 + lhalf * 16];
                s0 = WMMA_BF16(qf0, k00, s0);
                s0 = WMMA_BF16(qf1, k01, s0);
                s1 = WMMA_BF16(qf0, k10, s1);
                s1 = WMMA_BF16(qf1, k11, s1);
            }

            // ---- causal mask + online softmax ----
            float p0[8], p1[8];
#pragma unroll
            for (int r = 0; r < 8; ++r) {
                const int qg = qbase + r + lhalf * 8;
                const int k0 = jb + lmod;
                const int k1 = jb + 16 + lmod;
                float v0 = (k0 <= qg) ? s0[r] : MASKED_BIAS;
                float v1 = (k1 <= qg) ? s1[r] : MASKED_BIAS;
                float rmx = rowmax16(fmaxf(v0, v1));
                float nm  = fmaxf(mrow[r], rmx);
                float sc  = __expf(mrow[r] - nm);
                mrow[r] = nm;
                scl[r]  = sc;
                lrow[r] *= sc;
                p0[r] = __expf(v0 - nm);
                p1[r] = __expf(v1 - nm);
            }

            // ---- stage P (16x32 bf16) through LDS: C layout -> A fragment ----
#pragma unroll
            for (int r = 0; r < 8; ++r) {
                const int m = r + lhalf * 8;
                plds[wave][m][lmod]      = (bf16_t)p0[r];
                plds[wave][m][16 + lmod] = (bf16_t)p1[r];
            }
            const v16bf pa = *(const v16bf*)(&plds[wave][lmod][lhalf * 16]);

            // ---- row-sum via WMMA against a ones B-matrix ----
            v8f rs = WMMA_BF16(pa, ones, z);
#pragma unroll
            for (int r = 0; r < 8; ++r) lrow[r] += rs[r];

            // ---- O = O*scale + P*V ----
#pragma unroll
            for (int dt = 0; dt < 4; ++dt) {
                v16bf vf = *(const v16bf*)&vbuf[cur][dt * 16 + lmod][lhalf * 16];
#pragma unroll
                for (int r = 0; r < 8; ++r) o[dt][r] *= scl[r];
                o[dt] = WMMA_BF16(pa, vf, o[dt]);
            }
        }
    }

    // ---- finalize: O / l, write bf16 into [t*B+b][E] token-major buffer ----
    float linv[8];
#pragma unroll
    for (int r = 0; r < 8; ++r) linv[r] = 1.0f / lrow[r];
#pragma unroll
    for (int dt = 0; dt < 4; ++dt) {
#pragma unroll
        for (int r = 0; r < 8; ++r) {
            const int t = qbase + r + lhalf * 8;
            AO[((size_t)t * B_ + b) * E_ + h * DH_ + dt * 16 + lmod] =
                (bf16_t)(o[dt][r] * linv[r]);
        }
    }
}

// ---------------------------------------------------------------------------
// Output GEMM: out[M,E] = attn[M,E] * out_w[E,E]^T + out_b   (f32 output)
// ---------------------------------------------------------------------------
__global__ void __launch_bounds__(256) gemm_out(
    const bf16_t* __restrict__ A,     // [M, E] bf16
    const bf16_t* __restrict__ W,     // [E, E] bf16
    const float*  __restrict__ bias,  // [E]
    float* __restrict__ out)          // [M, E] f32
{
    const int lane  = threadIdx.x & 31;
    const int wave  = threadIdx.x >> 5;
    const int lmod  = lane & 15;
    const int lhalf = lane >> 4;

    const int mtile = blockIdx.x * 8 + wave;   // 0..511
    const int mbase = mtile * 16;
    const int nbase = blockIdx.y * 64;         // 0..960

    v8f acc[4];
    v8f z = {};
    acc[0] = z; acc[1] = z; acc[2] = z; acc[3] = z;

    const bf16_t* arow = A + (size_t)(mbase + lmod) * E_ + lhalf * 16;
    const bf16_t* wp[4];
#pragma unroll
    for (int ns = 0; ns < 4; ++ns)
        wp[ns] = W + (size_t)(nbase + ns * 16 + lmod) * E_ + lhalf * 16;

    gemm_16x64_core(arow, wp, acc);

#pragma unroll
    for (int ns = 0; ns < 4; ++ns) {
        const int f  = nbase + ns * 16 + lmod;
        const float bv = bias[f];
#pragma unroll
        for (int r = 0; r < 8; ++r) {
            const int m = mbase + r + lhalf * 8;   // m == t*B + b == output row
            out[(size_t)m * E_ + f] = acc[ns][r] + bv;
        }
    }
}

// ---------------------------------------------------------------------------
extern "C" void kernel_launch(void* const* d_in, const int* in_sizes, int n_in,
                              void* d_out, int out_size, void* d_ws, size_t ws_size,
                              hipStream_t stream)
{
    const float* query = (const float*)d_in[0];   // [T,B,E]
    const float* qkv_w = (const float*)d_in[1];   // [3E,E]
    const float* qkv_b = (const float*)d_in[2];   // [3E]
    const float* out_w = (const float*)d_in[3];   // [E,E]
    const float* out_b = (const float*)d_in[4];   // [E]
    float* out = (float*)d_out;

    char* ws = (char*)d_ws;
    const size_t MB = 1u << 20;
    bf16_t* Abf  = (bf16_t*)(ws + 0 * MB);   // 16 MB  query bf16   [M,E]
    bf16_t* Wbf  = (bf16_t*)(ws + 16 * MB);  //  6 MB  qkv_w bf16   [3E,E]
    bf16_t* OWbf = (bf16_t*)(ws + 22 * MB);  //  2 MB  out_w bf16   [E,E]
    bf16_t* Qh   = (bf16_t*)(ws + 24 * MB);  // 16 MB  Q head-major [BH,T,DH]
    bf16_t* Kh   = (bf16_t*)(ws + 40 * MB);  // 16 MB  K head-major [BH,T,DH]
    bf16_t* Vt   = (bf16_t*)(ws + 56 * MB);  // 16 MB  V transposed [BH,DH,T]
    bf16_t* AO   = (bf16_t*)(ws + 72 * MB);  // 16 MB  attn out     [M,E]

    cvt_f32_bf16<<<2048, 256, 0, stream>>>(query, Abf, M_ * E_);
    cvt_f32_bf16<<<2048, 256, 0, stream>>>(qkv_w, Wbf, 3 * E_ * E_);
    cvt_f32_bf16<<<1024, 256, 0, stream>>>(out_w, OWbf, E_ * E_);

    gemm_qkv  <<<dim3(64, 48), 256, 0, stream>>>(Abf, Wbf, qkv_b, Qh, Kh, Vt);
    attn_fused<<<dim3(128, 16), 128, 0, stream>>>(Qh, Kh, Vt, AO);
    gemm_out  <<<dim3(64, 16), 256, 0, stream>>>(AO, OWbf, out_b, out);
}
